// BiRNN_IFBU_84636625535323
// MI455X (gfx1250) — compile-verified, matrix operand order checked by reading, not run on hardware
//
#include <hip/hip_runtime.h>
#include <hip/hip_bf16.h>

// ---------------------------------------------------------------------------
// BiLSTM (3 layers, B=64, T=1024, U=512) + heads, CDNA5 / gfx1250.
// Strategy: persistent recurrent kernel per layer, bf16 WMMA 16x16x32 with f32
// accumulate, weights streamed from L2 (all fit in 192MB), c/h state in VGPRs,
// A = [x_t | h_prev] staged in LDS, device-wide split barrier per timestep.
// ---------------------------------------------------------------------------

#define T_LEN  1024
#define B_SZ   64
#define UNITS_ 512

typedef __attribute__((ext_vector_type(16))) __bf16 v16bf;
typedef __attribute__((ext_vector_type(8)))  float  v8f;

union ABfrag { uint4 q[2]; v16bf v; };

__device__ __forceinline__ unsigned short f2bf(float f) {
  unsigned int u = __float_as_uint(f);
  u += 0x7FFFu + ((u >> 16) & 1u);            // round-to-nearest-even
  return (unsigned short)(u >> 16);
}
__device__ __forceinline__ float bf2f(unsigned short h) {
  return __uint_as_float(((unsigned int)h) << 16);
}
__device__ __forceinline__ float sigm(float x) { return 1.0f / (1.0f + __expf(-x)); }

// ---------------------------------------------------------------------------
__global__ void init_bar_kernel(int* bar) {
  if (threadIdx.x < 16) bar[threadIdx.x] = 0;
}

__global__ void cast_bf16_kernel(const float* __restrict__ src,
                                 unsigned short* __restrict__ dst, int n) {
  for (int i = blockIdx.x * blockDim.x + threadIdx.x; i < n;
       i += gridDim.x * blockDim.x)
    dst[i] = f2bf(src[i]);
}

// Pack [Wk (din x 2048) ; Wr (512 x 2048)] into transposed bf16: Wt[n][k],
// n in [0,2048), k in [0, din+512). B-fragment loads become 2x16B per lane.
__global__ void pack_weights_kernel(const float* __restrict__ Wk,
                                    const float* __restrict__ Wr,
                                    unsigned short* __restrict__ Wt,
                                    int din, int Ktot) {
  int total = 2048 * Ktot;
  for (int i = blockIdx.x * blockDim.x + threadIdx.x; i < total;
       i += gridDim.x * blockDim.x) {
    int n = i / Ktot, k = i - n * Ktot;
    float v = (k < din) ? Wk[(size_t)k * 2048 + n]
                        : Wr[(size_t)(k - din) * 2048 + n];
    Wt[(size_t)n * Ktot + k] = f2bf(v);
  }
}

// Device-wide split barrier over gridDim.x workgroups (all resident).
__device__ __forceinline__ void grid_barrier(int* bar, int nwg, int target) {
  __syncthreads();
  __threadfence();
  if (threadIdx.x == 0) {
    int prev = __hip_atomic_fetch_add(&bar[0], 1, __ATOMIC_ACQ_REL,
                                      __HIP_MEMORY_SCOPE_AGENT);
    if (prev == nwg - 1) {
      __hip_atomic_store(&bar[0], 0, __ATOMIC_RELAXED, __HIP_MEMORY_SCOPE_AGENT);
      __hip_atomic_store(&bar[1], target, __ATOMIC_RELEASE,
                         __HIP_MEMORY_SCOPE_AGENT);
    } else {
      while (__hip_atomic_load(&bar[1], __ATOMIC_ACQUIRE,
                               __HIP_MEMORY_SCOPE_AGENT) < target)
        __builtin_amdgcn_s_sleep(1);
    }
  }
  __syncthreads();
}

// ---------------------------------------------------------------------------
// One BiLSTM layer, both directions concurrently.
// grid = 16 WGs: blockIdx/8 = direction, blockIdx%8 = unit-slice (64 units).
// block = 256 threads = 8 wave32s; wave>>1 = M-tile (16 batch rows),
// wave&1 = which 32-unit half. Each wave: 8 accum tiles = gates{i,f,g,o} x 2.
// ---------------------------------------------------------------------------
__global__ __launch_bounds__(256, 1) void lstm_layer_kernel(
    const unsigned short* __restrict__ Ain,   // [B][T][din] bf16
    int din,
    const unsigned short* __restrict__ WtF,   // [2048][din+512] bf16
    const unsigned short* __restrict__ WtB,
    const float* __restrict__ biasF,          // [2048]
    const float* __restrict__ biasB,
    const int* __restrict__ xmask,            // [B][T]
    unsigned short* __restrict__ hbuf,        // [2 dirs][2 bufs][64*512] bf16
    unsigned short* __restrict__ actOut,      // [B][T][1024] bf16
    int* __restrict__ bar) {
  const int Ktot = din + UNITS_;
  const int dir = blockIdx.x >> 3;
  const int wg  = blockIdx.x & 7;
  const int u0  = wg * 64;
  const unsigned short* Wt   = dir ? WtB : WtF;
  const float*          bias = dir ? biasB : biasF;
  unsigned short* hb = hbuf + (size_t)dir * 2 * B_SZ * UNITS_;

  const int tid  = threadIdx.x;
  const int wave = tid >> 5, lane = tid & 31;
  const int lo = lane & 15, hi = lane >> 4;
  const int mtile = wave >> 1;     // batch rows mtile*16 .. +16
  const int nh    = wave & 1;      // unit half: u0 + nh*32 + ut*16 + lo

  __shared__ __align__(16) unsigned short Atile[64 * 64];  // 8KB k-chunk stage
  __shared__ int msk[64];

  // zero h0 (buffer 0) for this WG's unit slice
  for (int i = tid; i < 64 * 64; i += 256) {
    int b = i >> 6, u = u0 + (i & 63);
    hb[b * UNITS_ + u] = 0;
  }
  int target = 1;
  grid_barrier(bar, gridDim.x, target++);

  float c[2][8], hp[2][8];
#pragma unroll
  for (int ut = 0; ut < 2; ++ut)
#pragma unroll
    for (int r = 0; r < 8; ++r) { c[ut][r] = 0.0f; hp[ut][r] = 0.0f; }

  float bv[4][2];
#pragma unroll
  for (int g = 0; g < 4; ++g)
#pragma unroll
    for (int ut = 0; ut < 2; ++ut)
      bv[g][ut] = bias[g * UNITS_ + u0 + nh * 32 + ut * 16 + lo];

  for (int t = 0; t < T_LEN; ++t) {
    const int tt  = dir ? (T_LEN - 1 - t) : t;
    const int cur = t & 1, nxt = cur ^ 1;
    const unsigned short* hcur = hb + cur * B_SZ * UNITS_;
    unsigned short*       hnxt = hb + nxt * B_SZ * UNITS_;

    if (tid < 64) msk[tid] = (xmask[tid * T_LEN + tt] != 1);

    v8f acc[4][2];
#pragma unroll
    for (int g = 0; g < 4; ++g)
#pragma unroll
      for (int ut = 0; ut < 2; ++ut)
#pragma unroll
        for (int r = 0; r < 8; ++r) acc[g][ut][r] = bv[g][ut];

    // z = [x_t | h_prev] @ [Wk;Wr] + b ; K-chunks of 64 (two WMMA K=32 steps)
    for (int kc = 0; kc < Ktot; kc += 64) {
      __syncthreads();
      {  // stage A chunk: 64 rows x 64 k, 256 threads x 32B
        int row  = tid >> 2;
        int koff = (tid & 3) << 4;
        int kg   = kc + koff;
        const unsigned short* src =
            (kg < din) ? (Ain + ((size_t)row * T_LEN + tt) * din + kg)
                       : (hcur + row * UNITS_ + (kg - din));
        uint4 q0 = *(const uint4*)(src);
        uint4 q1 = *(const uint4*)(src + 8);
        *(uint4*)&Atile[row * 64 + koff]     = q0;
        *(uint4*)&Atile[row * 64 + koff + 8] = q1;
      }
      __syncthreads();
#pragma unroll
      for (int kk = 0; kk < 64; kk += 32) {
        ABfrag af;  // ISA 16-bit A layout: lane lo = row, hi selects K groups
        const unsigned short* ar = &Atile[(mtile * 16 + lo) * 64 + kk];
        af.q[0] = *(const uint4*)(ar + hi * 8);
        af.q[1] = *(const uint4*)(ar + 16 + hi * 8);
#pragma unroll
        for (int g = 0; g < 4; ++g)
#pragma unroll
          for (int ut = 0; ut < 2; ++ut) {
            int ncol = g * UNITS_ + u0 + nh * 32 + ut * 16 + lo;
            const unsigned short* bp = Wt + (size_t)ncol * Ktot + kc + kk;
            __builtin_prefetch(bp + 64, 0, 1);  // next chunk, L2-resident
            ABfrag bf;
            bf.q[0] = *(const uint4*)(bp + hi * 8);
            bf.q[1] = *(const uint4*)(bp + 16 + hi * 8);
            acc[g][ut] = __builtin_amdgcn_wmma_f32_16x16x32_bf16(
                false, af.v, false, bf.v, (short)0, acc[g][ut], false, false);
          }
      }
    }
    __syncthreads();

    // gates + state update (fully in-register; D layout: M = hi*8+r, N = lo)
#pragma unroll
    for (int ut = 0; ut < 2; ++ut) {
      const int u = u0 + nh * 32 + ut * 16 + lo;
#pragma unroll
      for (int r = 0; r < 8; ++r) {
        const int b = mtile * 16 + hi * 8 + r;
        float iv = sigm(acc[0][ut][r]);
        float fv = sigm(acc[1][ut][r]);
        float gv = tanhf(acc[2][ut][r]);
        float ov = sigm(acc[3][ut][r]);
        float cn = fv * c[ut][r] + iv * gv;
        float hn = ov * tanhf(cn);
        if (msk[b]) { c[ut][r] = cn; hp[ut][r] = hn; }
        unsigned short h16 = f2bf(hp[ut][r]);
        hnxt[b * UNITS_ + u] = h16;
        actOut[((size_t)b * T_LEN + tt) * 1024 + dir * UNITS_ + u] = h16;
      }
    }
    grid_barrier(bar, gridDim.x, target++);
  }
}

// ---------------------------------------------------------------------------
__global__ void head_kernel(const unsigned short* __restrict__ h,   // [rows][1024]
                            const float* __restrict__ ppiW,         // [1024][2]
                            const float* __restrict__ ppib,
                            const float* __restrict__ burW,         // [1024]
                            const float* __restrict__ burb,
                            float* __restrict__ out, int rows) {
  int row = blockIdx.x * blockDim.x + threadIdx.x;
  if (row >= rows) return;
  const unsigned short* hr = h + (size_t)row * 1024;
  float a0 = 0.f, a1 = 0.f, a2 = 0.f;
  for (int k = 0; k < 1024; ++k) {
    float v = bf2f(hr[k]);
    v = v > 0.f ? v : 0.f;                       // relu
    a0 += v * ppiW[k * 2 + 0];
    a1 += v * ppiW[k * 2 + 1];
    a2 += v * burW[k];
  }
  a0 += ppib[0]; a1 += ppib[1];
  float mx = fmaxf(a0, a1);
  float e0 = __expf(a0 - mx), e1 = __expf(a1 - mx);
  float s = e0 + e1;
  out[(size_t)row * 2 + 0] = e0 / s;
  out[(size_t)row * 2 + 1] = e1 / s;
  out[(size_t)rows * 2 + row] = a2 + burb[0];
}

// ---------------------------------------------------------------------------
extern "C" void kernel_launch(void* const* d_in, const int* in_sizes, int n_in,
                              void* d_out, int out_size, void* d_ws,
                              size_t ws_size, hipStream_t stream) {
  const int B = B_SZ, T = T_LEN, DIN = 256;
  (void)in_sizes; (void)n_in; (void)out_size; (void)ws_size;

  const float* x  = (const float*)d_in[0];
  const int*   xm = (const int*)d_in[1];
  // d_in[2] = training (unused in eval path)
  const float* ppiW = (const float*)d_in[21];
  const float* ppib = (const float*)d_in[22];
  const float* burW = (const float*)d_in[23];
  const float* burb = (const float*)d_in[24];

  // workspace carve-out
  char*  w   = (char*)d_ws;
  size_t off = 0;
  auto alloc = [&](size_t bytes) {
    size_t o = off;
    off += (bytes + 255) & ~(size_t)255;
    return o;
  };
  unsigned short* xbf  = (unsigned short*)(w + alloc((size_t)B * T * DIN * 2));
  unsigned short* actA = (unsigned short*)(w + alloc((size_t)B * T * 1024 * 2));
  unsigned short* actB = (unsigned short*)(w + alloc((size_t)B * T * 1024 * 2));
  unsigned short* Wt[3][2];
  for (int l = 0; l < 3; ++l) {
    int Ktot = (l == 0 ? 256 : 1024) + UNITS_;
    for (int d = 0; d < 2; ++d)
      Wt[l][d] = (unsigned short*)(w + alloc((size_t)2048 * Ktot * 2));
  }
  unsigned short* hbuf = (unsigned short*)(w + alloc((size_t)2 * 2 * B * UNITS_ * 2));
  int* bar = (int*)(w + alloc(64));

  init_bar_kernel<<<1, 64, 0, stream>>>(bar);

  int nCast = B * T * DIN;
  cast_bf16_kernel<<<4096, 256, 0, stream>>>(x, xbf, nCast);

  for (int l = 0; l < 3; ++l) {
    int din = (l == 0) ? 256 : 1024;
    int Ktot = din + UNITS_;
    for (int d = 0; d < 2; ++d) {
      const float* Wk = (const float*)d_in[3 + l * 6 + d * 3 + 0];
      const float* Wr = (const float*)d_in[3 + l * 6 + d * 3 + 1];
      pack_weights_kernel<<<4096, 256, 0, stream>>>(Wk, Wr, Wt[l][d], din, Ktot);
    }
  }

  // layer 0: xbf -> actA ; layer 1: actA -> actB ; layer 2: actB -> actA
  const unsigned short* ins[3]  = {xbf, actA, actB};
  unsigned short*       outs[3] = {actA, actB, actA};
  for (int l = 0; l < 3; ++l) {
    int din = (l == 0) ? 256 : 1024;
    const float* bF = (const float*)d_in[3 + l * 6 + 0 * 3 + 2];
    const float* bB = (const float*)d_in[3 + l * 6 + 1 * 3 + 2];
    lstm_layer_kernel<<<16, 256, 0, stream>>>(ins[l], din, Wt[l][0], Wt[l][1],
                                              bF, bB, xm, hbuf, outs[l],
                                              bar + l * 2);
  }

  int rows = B * T;
  head_kernel<<<(rows + 255) / 256, 256, 0, stream>>>(actA, ppiW, ppib, burW,
                                                      burb, (float*)d_out, rows);
}
